// InfinityRNNLayer_28853590295064
// MI455X (gfx1250) — matrix operand order repaired and checked by reading, not compiled, not run on hardware
//
#include <hip/hip_runtime.h>

// ---------------- problem constants ----------------
#define SEQ    2048
#define BATCH  8
#define DIM    512
#define HIDN   512
#define NNODE  7
#define MTOT   (SEQ * BATCH)      // 16384 GEMM rows
#define N3     (3 * HIDN)         // 1536 GEMM cols
#define NCHAN  (BATCH * HIDN)     // 4096 scan channels
#define CHUNK  32
#define NCHUNK (SEQ / CHUNK)      // 64

// ---------------- types ----------------
typedef __attribute__((ext_vector_type(16))) __bf16        v16bf;
typedef __attribute__((ext_vector_type(8)))  float         v8f;
typedef __attribute__((ext_vector_type(4)))  unsigned int  u32x4;
typedef __attribute__((ext_vector_type(8)))  int           i32x8;
typedef __attribute__((ext_vector_type(4)))  int           i32x4;

union FragBF { v16bf v; u32x4 u[2]; };

#if defined(__HIP_DEVICE_COMPILE__) && __has_builtin(__builtin_amdgcn_tensor_load_to_lds) && \
    __has_builtin(__builtin_amdgcn_s_wait_tensorcnt)
#define USE_TDM 1
#endif

__device__ __forceinline__ float sigmoidf(float x) {
  return 1.0f / (1.0f + expf(-x));
}

__device__ __forceinline__ unsigned short f2bf(float f) {
  unsigned int u = __float_as_uint(f);
  unsigned int r = (u + 0x7FFFu + ((u >> 16) & 1u)) >> 16;  // RNE
  return (unsigned short)r;
}

// ---------------- fp32 -> bf16 conversion (X) ----------------
__global__ void cvt_bf16_kernel(const float* __restrict__ in,
                                unsigned short* __restrict__ out, size_t n) {
  size_t i = (size_t)blockIdx.x * blockDim.x + threadIdx.x;
  if (i < n) out[i] = f2bf(in[i]);
}

// ---------------- W transpose+convert: WhT[node][n][k] = bf16(W[node][k][n]) --------
__global__ void __launch_bounds__(256)
wtrans_kernel(const float* __restrict__ W, unsigned short* __restrict__ WhT) {
  __shared__ float sh[32][33];
  const int node = blockIdx.z;
  const int n0 = blockIdx.x * 32;
  const int k0 = blockIdx.y * 32;
  const int tx = threadIdx.x & 31;
  const int ty = threadIdx.x >> 5;   // 0..7
#pragma unroll
  for (int j = 0; j < 4; ++j) {
    int k = k0 + ty + j * 8;
    sh[ty + j * 8][tx] = W[((size_t)node * DIM + k) * N3 + n0 + tx];
  }
  __syncthreads();
#pragma unroll
  for (int j = 0; j < 4; ++j) {
    int n = n0 + ty + j * 8;
    WhT[((size_t)node * N3 + n) * DIM + k0 + tx] = f2bf(sh[tx][ty + j * 8]);
  }
}

// ---------------- gate projection: t = sigmoid(X @ W_T[i] + b_T[i]) ----------------
__global__ void __launch_bounds__(256)
tproj_kernel(const float* __restrict__ X, const float* __restrict__ W_T,
             const float* __restrict__ b_T, float* __restrict__ T) {
  int m    = blockIdx.x * 256 + threadIdx.x;   // 0..MTOT-1
  int node = blockIdx.y;
  const float* x = X + (size_t)m * DIM;
  const float* w = W_T + (size_t)node * DIM * 2;
  float a0 = 0.0f, a1 = 0.0f;
#pragma unroll 8
  for (int d = 0; d < DIM; d += 2) {
    float2 xv = *(const float2*)(x + d);
    float4 wv = *(const float4*)(w + (size_t)d * 2);
    a0 += xv.x * wv.x + xv.y * wv.z;
    a1 += xv.x * wv.y + xv.y * wv.w;
  }
  a0 = sigmoidf(a0 + b_T[node * 2 + 0]);
  a1 = sigmoidf(a1 + b_T[node * 2 + 1]);
  float* dst = T + ((size_t)node * MTOT + m) * 2;
  dst[0] = a0; dst[1] = a1;
}

// ---------------- bf16 WMMA GEMM: V[M,1536] = Xh[M,512] * WhT[1536,512]^T ----------
// 128x128 tile per 256-thread block (8 waves, each wave 32x64 = 2x4 WMMA tiles).
// Both A and B tiles are 128 rows x 64 k, staged row-major into LDS (stride 72).
// TDM path: double-buffered LDS, Tensor Data Mover streams stage k+1 while
// the waves run WMMAs on stage k (s_wait_tensorcnt <= 2 keeps 1 pair in flight).
#define TILE_M 128
#define TILE_N 128
#define KBLK   64
#define LDA    72   // padded row stride (bf16 elems): 64 data + 8 pad
#define LDB    72
#define KITER  (DIM / KBLK)

#ifdef USE_TDM
// Tensor Data Mover: 2D tile load (128 rows x 64 bf16 cols) with LDS row padding.
// D# per CDNA5 ISA ch.8: group0 = {count, lds_addr, global_addr, type=2},
// group1 = {data_size=2B, pad 4 DWORDs per 32 DWORDs (row stride 72 bf16),
//           tensor dims, tile 64x128, dim0 stride}.
__device__ __forceinline__ void tdm_load_tile_2d(unsigned lds_off,
                                                 const unsigned short* gptr,
                                                 unsigned tensor_d0,
                                                 unsigned tensor_d1) {
  unsigned long long ga = (unsigned long long)(size_t)gptr;
  u32x4 g0 = { 1u,                                  // count=1 (valid descriptor)
               lds_off,                             // lds_addr (bytes)
               (unsigned)ga,                        // global_addr[31:0]
               (unsigned)(ga >> 32) | (2u << 30) }; // global_addr[56:32] | type=2
  int w0 = (int)((1u << 16)    // data_size = 1 -> 2 bytes
               | (1u << 20)    // pad_enable
               | (4u << 22)    // pad_interval: every 32 DWORDs (= one 128B row)
               | (3u << 25));  // pad_amount: 4 DWORDs (8 bf16) -> stride 72
  i32x8 g1 = { w0,
               (int)((tensor_d0 & 0xFFFFu) << 16),                          // dim0 lo
               (int)(((tensor_d0 >> 16) & 0xFFFFu) | ((tensor_d1 & 0xFFFFu) << 16)),
               (int)(((tensor_d1 >> 16) & 0xFFFFu) | (64u << 16)),          // tile_dim0=64
               128,                                                          // tile_dim1=128
               (int)tensor_d0,                                               // dim0 stride
               0, 0 };
  i32x4 z4 = {0, 0, 0, 0};
#if __clang_major__ >= 23
  i32x8 z8 = {0, 0, 0, 0, 0, 0, 0, 0};
  __builtin_amdgcn_tensor_load_to_lds(g0, g1, z4, z4, z8, 0);
#else
  __builtin_amdgcn_tensor_load_to_lds(g0, g1, z4, z4, 0);
#endif
}
#endif  // USE_TDM

__global__ void __launch_bounds__(256)
gemm_bf16_wmma_kernel(const unsigned short* __restrict__ Xh,
                      const unsigned short* __restrict__ WhT,
                      float* __restrict__ V) {
#ifdef USE_TDM
  __shared__ __align__(16) unsigned short As[2][TILE_M * LDA];
  __shared__ __align__(16) unsigned short Bs[2][TILE_N * LDB];
#else
  __shared__ __align__(16) unsigned short As[1][TILE_M * LDA];
  __shared__ __align__(16) unsigned short Bs[1][TILE_N * LDB];
#endif

  const int t    = threadIdx.x;
  const int lane = t & 31;
  const int wid  = t >> 5;
  const int wm   = wid & 3;   // wave M quadrant: rows wm*32..wm*32+31
  const int wn   = wid >> 2;  // wave N half:    cols wn*64..wn*64+63
  const int m0   = blockIdx.x * TILE_M;
  const int n0   = blockIdx.y * TILE_N;

  v8f acc[2][4];
#pragma unroll
  for (int i = 0; i < 2; ++i)
#pragma unroll
    for (int j = 0; j < 4; ++j)
      acc[i][j] = (v8f){0.f, 0.f, 0.f, 0.f, 0.f, 0.f, 0.f, 0.f};

#ifdef USE_TDM
  // prologue: kick off stage 0 into buffer 0
  if (wid == 0) {
    tdm_load_tile_2d((unsigned)(size_t)&As[0][0], Xh  + (size_t)m0 * DIM, DIM, MTOT);
    tdm_load_tile_2d((unsigned)(size_t)&Bs[0][0], WhT + (size_t)n0 * DIM, DIM, N3);
  }
#endif

  for (int it = 0; it < KITER; ++it) {
    const int kk = it * KBLK;
#ifdef USE_TDM
    const int cur = it & 1;
    const int nxt = cur ^ 1;
    if (wid == 0) {
      if (it + 1 < KITER) {
        // stream next stage while this stage computes; consumers of buf[nxt]
        // finished at the previous end-of-iteration barrier.
        tdm_load_tile_2d((unsigned)(size_t)&As[nxt][0],
                         Xh + (size_t)m0 * DIM + kk + KBLK, DIM, MTOT);
        tdm_load_tile_2d((unsigned)(size_t)&Bs[nxt][0],
                         WhT + (size_t)n0 * DIM + kk + KBLK, DIM, N3);
        __builtin_amdgcn_s_wait_tensorcnt(2);  // in-order: cur pair has landed
      } else {
        __builtin_amdgcn_s_wait_tensorcnt(0);
      }
    }
    const unsigned short* Ap = &As[cur][0];
    const unsigned short* Bp = &Bs[cur][0];
#else
    // manual staging: both tiles are 128 rows x 64 bf16, contiguous b128 copies
#pragma unroll
    for (int p = 0; p < 4; ++p) {
      int flat = (p * 256 + t) * 8;
      int row  = flat >> 6;
      int col  = flat & 63;
      u32x4 va = *(const u32x4*)(Xh + (size_t)(m0 + row) * DIM + kk + col);
      *(u32x4*)(&As[0][0] + row * LDA + col) = va;
      u32x4 vb = *(const u32x4*)(WhT + (size_t)(n0 + row) * DIM + kk + col);
      *(u32x4*)(&Bs[0][0] + row * LDB + col) = vb;
    }
    if (kk + KBLK < DIM) {
      __builtin_prefetch(Xh + (size_t)(m0 + (t >> 1)) * DIM + kk + KBLK, 0, 1);
      __builtin_prefetch(WhT + (size_t)(n0 + (t >> 1)) * DIM + kk + KBLK, 0, 1);
    }
    const unsigned short* Ap = &As[0][0];
    const unsigned short* Bp = &Bs[0][0];
#endif
    __syncthreads();

#pragma unroll
    for (int ks = 0; ks < KBLK; ks += 32) {
      FragBF a[2];
#pragma unroll
      for (int mt = 0; mt < 2; ++mt) {
        // A layout (16x32 bf16): lanes 0-15 = rows, half-select by lane>>4:
        //   elems 0..7 = K[kh..kh+7], elems 8..15 = K[16+kh..23+kh], kh=(lane>>4)*8
        int row = wm * 32 + mt * 16 + (lane & 15);
        int kh  = (lane >> 4) * 8;
        a[mt].u[0] = *(const u32x4*)(Ap + row * LDA + ks + kh);
        a[mt].u[1] = *(const u32x4*)(Ap + row * LDA + ks + 16 + kh);
      }
      FragBF b[4];
#pragma unroll
      for (int nt = 0; nt < 4; ++nt) {
        // B layout (32x16 bf16): lane&15 = column, elems 0..7 = K[kh..kh+7], elems 8..15 = K[16+kh..23+kh], kh=(lane>>4)*8
        int ncol  = wn * 64 + nt * 16 + (lane & 15);
        int kbase = ks + (lane >> 4) * 8;
        b[nt].u[0] = *(const u32x4*)(Bp + ncol * LDB + kbase);
        b[nt].u[1] = *(const u32x4*)(Bp + ncol * LDB + kbase + 16);
      }
#pragma unroll
      for (int mt = 0; mt < 2; ++mt)
#pragma unroll
        for (int nt = 0; nt < 4; ++nt)
          acc[mt][nt] = __builtin_amdgcn_wmma_f32_16x16x32_bf16(
              false, a[mt].v, false, b[nt].v, (short)0, acc[mt][nt], false, false);
    }
    __syncthreads();
  }

  // --- epilogue: C/D layout — VGPR r: M = (lane>>4)*8 + r, N = lane&15 ---
#pragma unroll
  for (int mt = 0; mt < 2; ++mt) {
#pragma unroll
    for (int nt = 0; nt < 4; ++nt) {
      int col     = n0 + wn * 64 + nt * 16 + (lane & 15);
      int rowbase = m0 + wm * 32 + mt * 16 + (lane >> 4) * 8;
#pragma unroll
      for (int r = 0; r < 8; ++r)
        V[(size_t)(rowbase + r) * N3 + col] = acc[mt][nt][r];
    }
  }
}

// ---------------- chunked linear-recurrence scan ----------------
// c_t = f*z + (1-f)*c  ==>  over a chunk: c_out = P*c_in + Q, P = prod(1-f).
__global__ void __launch_bounds__(256)
phase1_kernel(const float* __restrict__ V, const float* __restrict__ Tn,
              const float* __restrict__ Hf, float* __restrict__ P,
              float* __restrict__ Q, int hasChild) {
  int ch    = blockIdx.y * 256 + threadIdx.x;   // 0..4095  (= b*512 + h)
  int chunk = blockIdx.x;
  int b = ch >> 9, h = ch & 511;
  int s0 = chunk * CHUNK;
  size_t vF = (size_t)(s0 * BATCH + b) * N3 + HIDN + h;  // F zone
  size_t vZ = (size_t)(s0 * BATCH + b) * N3 + h;         // Z zone
  float p = 1.0f, q = 0.0f;
#pragma unroll 4
  for (int s = s0; s < s0 + CHUNK; ++s) {
    float fraw = V[vF];
    if (hasChild) {
      float t1 = Tn[(size_t)(s * BATCH + b) * 2];
      float f2 = Hf[(size_t)s * NCHAN + ch];
      fraw = f2 * t1 + (1.0f - t1) * fraw;
    }
    float f   = sigmoidf(fraw);
    float z   = fmaxf(V[vZ], 0.0f);
    float omf = 1.0f - f;
    q = f * z + omf * q;
    p *= omf;
    vF += (size_t)BATCH * N3;
    vZ += (size_t)BATCH * N3;
  }
  P[(size_t)chunk * NCHAN + ch] = p;
  Q[(size_t)chunk * NCHAN + ch] = q;
}

__global__ void __launch_bounds__(256)
phase2_kernel(const float* __restrict__ P, const float* __restrict__ Q,
              float* __restrict__ Cin) {
  int ch = blockIdx.x * 256 + threadIdx.x;
  float c = 0.0f;
  for (int k = 0; k < NCHUNK; ++k) {
    Cin[(size_t)k * NCHAN + ch] = c;
    c = P[(size_t)k * NCHAN + ch] * c + Q[(size_t)k * NCHAN + ch];
  }
}

__global__ void __launch_bounds__(256)
phase3_kernel(const float* __restrict__ V, const float* __restrict__ Tn,
              const float* __restrict__ Hf, const float* __restrict__ Ho,
              const float* __restrict__ Cin, const float* __restrict__ X,
              float* __restrict__ Hout, float* __restrict__ ChunkMax,
              int hasChild, int isRoot) {
  int ch    = blockIdx.y * 256 + threadIdx.x;
  int chunk = blockIdx.x;
  int b = ch >> 9, h = ch & 511;
  int s0 = chunk * CHUNK;
  size_t vZ = (size_t)(s0 * BATCH + b) * N3 + h;
  float c  = Cin[(size_t)chunk * NCHAN + ch];
  float mx = -3.4e38f;
#pragma unroll 4
  for (int s = s0; s < s0 + CHUNK; ++s) {
    float fraw = V[vZ + HIDN];
    float oraw = V[vZ + 2 * HIDN];
    if (hasChild) {
      size_t ti = (size_t)(s * BATCH + b) * 2;
      float t1 = Tn[ti], t2 = Tn[ti + 1];
      size_t hi = (size_t)s * NCHAN + ch;
      fraw = Hf[hi] * t1 + (1.0f - t1) * fraw;
      oraw = Ho[hi] * t2 + (1.0f - t2) * oraw;
    }
    float f = sigmoidf(fraw);
    float z = fmaxf(V[vZ], 0.0f);
    c = f * z + (1.0f - f) * c;
    float hp = sigmoidf(oraw) * c;
    mx = fmaxf(mx, hp);
    Hout[(size_t)s * NCHAN + ch] = hp + X[(size_t)s * NCHAN + ch];
    vZ += (size_t)BATCH * N3;
  }
  if (isRoot) ChunkMax[(size_t)chunk * NCHAN + ch] = mx;
}

__global__ void __launch_bounds__(256)
rootmax_kernel(const float* __restrict__ ChunkMax, float* __restrict__ out) {
  int ch = blockIdx.x * 256 + threadIdx.x;
  float m = -3.4e38f;
  for (int k = 0; k < NCHUNK; ++k)
    m = fmaxf(m, ChunkMax[(size_t)k * NCHAN + ch]);
  out[ch] = m;
}

// ---------------- host orchestration ----------------
extern "C" void kernel_launch(void* const* d_in, const int* in_sizes, int n_in,
                              void* d_out, int out_size, void* d_ws, size_t ws_size,
                              hipStream_t stream) {
  const float* X      = (const float*)d_in[0];
  const float* W_proj = (const float*)d_in[1];
  const float* W_T    = (const float*)d_in[2];
  const float* b_T    = (const float*)d_in[3];
  float* out = (float*)d_out;

  // workspace carve-up (~335 MB)
  char* ws = (char*)d_ws;
  size_t off = 0;
  auto carve = [&](size_t bytes) -> void* {
    void* p = ws + off;
    off += (bytes + 255) & ~(size_t)255;
    return p;
  };
  unsigned short* Xh  = (unsigned short*)carve((size_t)MTOT * DIM * 2);
  unsigned short* WhT = (unsigned short*)carve((size_t)NNODE * N3 * DIM * 2);
  float* T   = (float*)carve((size_t)NNODE * MTOT * 2 * 4);
  float* V   = (float*)carve((size_t)MTOT * N3 * 4);          // reused per node
  float* Hb[NNODE];
  Hb[0] = nullptr;  // root writes straight to d_out
  for (int i = 1; i < NNODE; ++i)
    Hb[i] = (float*)carve((size_t)MTOT * HIDN * 4);
  float* P    = (float*)carve((size_t)NCHUNK * NCHAN * 4);
  float* Q    = (float*)carve((size_t)NCHUNK * NCHAN * 4);
  float* Cin  = (float*)carve((size_t)NCHUNK * NCHAN * 4);
  float* CMax = (float*)carve((size_t)NCHUNK * NCHAN * 4);

  // X -> bf16; W_proj -> transposed bf16 [node][n][k]
  {
    size_t nx = (size_t)MTOT * DIM;
    cvt_bf16_kernel<<<(unsigned)((nx + 255) / 256), 256, 0, stream>>>(X, Xh, nx);
  }
  wtrans_kernel<<<dim3(N3 / 32, DIM / 32, NNODE), 256, 0, stream>>>(W_proj, WhT);

  // gate projections for all nodes
  tproj_kernel<<<dim3(MTOT / 256, NNODE), 256, 0, stream>>>(X, W_T, b_T, T);

  // topological node order: leaves -> mid -> root; single V buffer reused
  const int order[NNODE] = {3, 4, 5, 6, 1, 2, 0};
  for (int oi = 0; oi < NNODE; ++oi) {
    int i = order[oi];
    gemm_bf16_wmma_kernel<<<dim3(MTOT / TILE_M, N3 / TILE_N), 256, 0, stream>>>(
        Xh, WhT + (size_t)i * N3 * DIM, V);

    int hasChild    = (i <= 2) ? 1 : 0;
    const float* Hf = hasChild ? Hb[2 * i + 1] : nullptr;
    const float* Ho = hasChild ? Hb[2 * i + 2] : nullptr;
    const float* Tn = T + (size_t)i * MTOT * 2;

    phase1_kernel<<<dim3(NCHUNK, NCHAN / 256), 256, 0, stream>>>(V, Tn, Hf, P, Q, hasChild);
    phase2_kernel<<<NCHAN / 256, 256, 0, stream>>>(P, Q, Cin);
    float* Hout = (i == 0) ? out : Hb[i];
    phase3_kernel<<<dim3(NCHUNK, NCHAN / 256), 256, 0, stream>>>(
        V, Tn, Hf, Ho, Cin, X, Hout, CMax, hasChild, (i == 0) ? 1 : 0);
  }
  rootmax_kernel<<<NCHAN / 256, 256, 0, stream>>>(CMax, out + (size_t)MTOT * HIDN);
}
